// GavinMultiHeadPerformerAttention_56538949484739
// MI455X (gfx1250) — matrix operand order repaired and checked by reading, not compiled, not run on hardware
//
#include <hip/hip_runtime.h>
#include <math.h>

// ---------------------------------------------------------------------------
// Performer attention for MI455X (gfx1250, wave32, WMMA bf16 16x16x32).
// All matmul operands pre-converted to bf16 once; hot loops stage tiles with
// async global->LDS DMA (CDNA5 ASYNCcnt path) or plain LDS transposes, then
// v_wmma_f32_16x16x32_bf16 with f32 accumulation.
// ---------------------------------------------------------------------------

typedef __attribute__((ext_vector_type(16))) __bf16        v16bf;
typedef __attribute__((ext_vector_type(2)))  __bf16        v2bf;
typedef __attribute__((ext_vector_type(8)))  float         v8f;
typedef __attribute__((ext_vector_type(8)))  unsigned int  v8u;

__device__ __forceinline__ unsigned int packbf(float lo, float hi) {
#if __has_builtin(__builtin_amdgcn_cvt_pk_bf16_f32)
  v2bf p = __builtin_amdgcn_cvt_pk_bf16_f32(lo, hi);
  return __builtin_bit_cast(unsigned int, p);
#else
  v2bf p; p[0] = (__bf16)lo; p[1] = (__bf16)hi;
  return __builtin_bit_cast(unsigned int, p);
#endif
}
__device__ __forceinline__ unsigned short f2bf(float f) {
  __bf16 b = (__bf16)f;
  return __builtin_bit_cast(unsigned short, b);
}
__device__ __forceinline__ float bf2f(unsigned short s) {
  unsigned int u = ((unsigned int)s) << 16;
  return __builtin_bit_cast(float, u);
}

// CDNA5 async global->LDS copy (8 bytes per lane), tracked by ASYNCcnt.
__device__ __forceinline__ void async_b64(unsigned lds_off, const void* g) {
  asm volatile("global_load_async_to_lds_b64 %0, %1, off"
               :: "v"(lds_off), "v"((unsigned long long)g) : "memory");
}
__device__ __forceinline__ void wait_async0() {
#if __has_builtin(__builtin_amdgcn_s_wait_asynccnt)
  __builtin_amdgcn_s_wait_asynccnt(0);
#else
  asm volatile("s_wait_asynccnt 0" ::: "memory");
#endif
}

// A fragment (16x32 bf16): lanes 0..15 = M rows, kbase = (r/4)*16+grp*8+(r%4)*2
__device__ __forceinline__ v16bf ld_a(const unsigned short* lds, int m0, int k0, int stride) {
  int lane = threadIdx.x & 31;
  int grp  = lane >> 4, mr = lane & 15;
  const unsigned short* row = lds + (m0 + mr) * stride + k0;
  v8u u;
#pragma unroll
  for (int r = 0; r < 8; ++r) {
    int kb = ((r >> 2) << 4) | (grp << 3) | ((r & 3) << 1);
    u[r] = *(const unsigned int*)(row + kb);
  }
  return __builtin_bit_cast(v16bf, u);
}
// B fragment from LDS stored transposed [n][k]; kbase = (r/4)*16+grp*8+(r%4)*2
__device__ __forceinline__ v16bf ld_b(const unsigned short* lds, int n0, int k0, int stride) {
  int lane = threadIdx.x & 31;
  int grp  = lane >> 4, nc = lane & 15;
  const unsigned short* row = lds + (n0 + nc) * stride + k0;
  v8u u;
#pragma unroll
  for (int r = 0; r < 8; ++r) u[r] = *(const unsigned int*)(row + (((r >> 2) << 4) | (grp << 3) | ((r & 3) << 1)));
  return __builtin_bit_cast(v16bf, u);
}
__device__ __forceinline__ v8f wmma_bf16(v16bf a, v16bf b, v8f c) {
  return __builtin_amdgcn_wmma_f32_16x16x32_bf16(false, a, false, b, (short)0, c, false, false);
}

// ---------------------------------------------------------------------------
// fp32 -> bf16 (optionally scaled) bulk converter; one float4 per thread.
// ---------------------------------------------------------------------------
__global__ __launch_bounds__(256)
void cvt_bf16_kernel(const float* __restrict__ in, unsigned short* __restrict__ out,
                     int n4, float scale) {
  int i = blockIdx.x * 256 + threadIdx.x;
  if (i >= n4) return;
  float4 v = ((const float4*)in)[i];
  uint2 o;
  o.x = packbf(v.x * scale, v.y * scale);
  o.y = packbf(v.z * scale, v.w * scale);
  ((uint2*)out)[i] = o;
}

// ---------------------------------------------------------------------------
// C[16384,1024] = A_bf16 @ W_bf16 + bias.  Block 64x64, 4 waves.  A tile is
// async-DMAed into LDS; W tile transposed to [n][k] via LDS scatter.
// ---------------------------------------------------------------------------
template <bool BF16OUT>
__global__ __launch_bounds__(128)
void gemm_bias_kernel(const unsigned short* __restrict__ A,
                      const unsigned short* __restrict__ W,
                      const float* __restrict__ bias,
                      float* __restrict__ Cf, unsigned short* __restrict__ Cb) {
  __shared__ __align__(16) unsigned short As[64 * 36];
  __shared__ __align__(16) unsigned short Ws[64 * 36];
  const int tid  = threadIdx.x;
  const int row0 = blockIdx.x * 64;
  const int col0 = blockIdx.y * 64;
  const unsigned lds_a = (unsigned)(size_t)(void*)As;
  const unsigned int* Wu = (const unsigned int*)W;
  v8f acc[4] = {};
  for (int k0 = 0; k0 < 1024; k0 += 32) {
#pragma unroll
    for (int i = 0; i < 4; ++i) {                 // A tile 64 rows x 64B, async
      int idx = tid + i * 128;
      int r = idx >> 3, ch = idx & 7;
      async_b64(lds_a + r * 72 + ch * 8,
                (const char*)A + ((size_t)(row0 + r) * 1024 + k0) * 2 + ch * 8);
    }
#pragma unroll
    for (int i = 0; i < 8; ++i) {                 // W tile 32x64 -> [n][k]
      int idx = tid + i * 128;
      int kk = idx >> 5, np = idx & 31;
      unsigned int u = Wu[(size_t)(k0 + kk) * 512 + (col0 >> 1) + np];
      Ws[(np * 2) * 36 + kk]     = (unsigned short)(u & 0xFFFFu);
      Ws[(np * 2 + 1) * 36 + kk] = (unsigned short)(u >> 16);
    }
    wait_async0();
    __syncthreads();
    int wv = tid >> 5;
    v16bf a = ld_a(As, wv * 16, 0, 36);
#pragma unroll
    for (int nt = 0; nt < 4; ++nt) {
      v16bf b = ld_b(Ws, nt * 16, 0, 36);
      acc[nt] = wmma_bf16(a, b, acc[nt]);
    }
    __syncthreads();
  }
  int wv = tid >> 5, lane = tid & 31, grp = lane >> 4, c = lane & 15;
  int rbase = row0 + wv * 16 + grp * 8;
#pragma unroll
  for (int nt = 0; nt < 4; ++nt) {
    int col = col0 + nt * 16 + c;
    float bi = bias[col];
#pragma unroll
    for (int r = 0; r < 8; ++r) {
      float v = acc[nt][r] + bi;
      if (BF16OUT) Cb[(size_t)(rbase + r) * 1024 + col] = f2bf(v);
      else         Cf[(size_t)(rbase + r) * 1024 + col] = v;
    }
  }
}

// ---------------------------------------------------------------------------
// phi: dash = x_bf16 @ (proj*dn)^T  [64x256, K=64] per (b,h,l-tile), with
// fused row-max / diag / exp epilogue.  Both tiles async-DMAed (pure copies).
// diag = 0.5*dn^2*sum(x^2) = 0.0625*sum(x^2).
// ---------------------------------------------------------------------------
__global__ __launch_bounds__(256)
void phi_kernel(const unsigned short* __restrict__ X,
                const unsigned short* __restrict__ projbf,
                unsigned short* __restrict__ out) {
  __shared__ __align__(16) unsigned short As[64 * 68];
  __shared__ __align__(16) unsigned short Ps[256 * 68];
  __shared__ float diag_l[64];
  __shared__ float wmax[2 * 64];
  const int tid = threadIdx.x;
  const int l0  = blockIdx.x * 64;
  const int h   = blockIdx.y;
  const int b   = blockIdx.z;
  const unsigned lds_a = (unsigned)(size_t)(void*)As;
  const unsigned lds_p = (unsigned)(size_t)(void*)Ps;
#pragma unroll
  for (int i = 0; i < 4; ++i) {                   // x tile 64 rows x 128B
    int idx = tid + i * 256;
    int r = idx >> 4, ch = idx & 15;
    async_b64(lds_a + r * 136 + ch * 8,
              (const char*)X + ((size_t)(b * 4096 + l0 + r) * 1024 + h * 64) * 2 + ch * 8);
  }
#pragma unroll
  for (int i = 0; i < 16; ++i) {                  // proj 256 rows x 128B
    int idx = tid + i * 256;
    int r = idx >> 4, ch = idx & 15;
    async_b64(lds_p + r * 136 + ch * 8, (const char*)projbf + (size_t)r * 128 + ch * 8);
  }
  wait_async0();
  __syncthreads();
  if (tid < 64) {                                 // diag
    float s = 0.f;
    for (int k = 0; k < 64; ++k) { float x = bf2f(As[tid * 68 + k]); s += x * x; }
    diag_l[tid] = 0.0625f * s;
  }
  const int wv = tid >> 5, lane = tid & 31, grp = lane >> 4, c = lane & 15;
  const int rt = wv >> 1, ch = wv & 1;
  const int m0 = rt * 16;
  v8f acc[8] = {};
#pragma unroll
  for (int ks = 0; ks < 2; ++ks) {
    v16bf a = ld_a(As, m0, ks * 32, 68);
#pragma unroll
    for (int j = 0; j < 8; ++j) {
      v16bf bb = ld_b(Ps, (ch * 8 + j) * 16, ks * 32, 68);
      acc[j] = wmma_bf16(a, bb, acc[j]);
    }
  }
  float mx[8];
#pragma unroll
  for (int r = 0; r < 8; ++r) {                   // row max over 128 cols
    float m = acc[0][r];
#pragma unroll
    for (int j = 1; j < 8; ++j) m = fmaxf(m, acc[j][r]);
#pragma unroll
    for (int s = 1; s < 16; s <<= 1) m = fmaxf(m, __shfl_xor(m, s, 32));
    mx[r] = m;
  }
  if (c == 0) {
#pragma unroll
    for (int r = 0; r < 8; ++r) wmax[ch * 64 + m0 + grp * 8 + r] = mx[r];
  }
  __syncthreads();
#pragma unroll
  for (int j = 0; j < 8; ++j) {
    int col = (ch * 8 + j) * 16 + c;
#pragma unroll
    for (int r = 0; r < 8; ++r) {
      int row = m0 + grp * 8 + r;
      float rm  = fmaxf(wmax[row], wmax[64 + row]);
      float val = 0.0625f * (expf(acc[j][r] - diag_l[row] - rm) + 1e-6f);
      out[(size_t)((b * 4096 + l0 + row) * 16 + h) * 256 + col] = f2bf(val);
    }
  }
}

// ---------------------------------------------------------------------------
// kv[b,h] = kp^T @ v  (256x64, K=4096) in bf16 out, plus deterministic
// ksum[b,h,m].  One block per (b,h), 8 waves; wave -> 2 m-tiles x 4 n-tiles.
// ---------------------------------------------------------------------------
__global__ __launch_bounds__(256)
void kv_ksum_kernel(const unsigned short* __restrict__ kp,
                    const unsigned short* __restrict__ vb,
                    unsigned short* __restrict__ kv, float* __restrict__ ksum) {
  __shared__ __align__(16) unsigned short As[256 * 36];
  __shared__ __align__(16) unsigned short Bs[64 * 36];
  const int tid = threadIdx.x;
  const int bh  = blockIdx.x;
  const int b = bh >> 4, h = bh & 15;
  const int wv = tid >> 5;
  v8f acc[8] = {};
  float ks = 0.f;
  for (int l0 = 0; l0 < 4096; l0 += 32) {
#pragma unroll
    for (int i = 0; i < 16; ++i) {                // kp tile -> [m][l] transpose
      int idx = tid + i * 256;
      int l = idx >> 7, mp = idx & 127;
      unsigned int u = *(const unsigned int*)(kp + (size_t)((b * 4096 + l0 + l) * 16 + h) * 256 + mp * 2);
      As[(mp * 2) * 36 + l]     = (unsigned short)(u & 0xFFFFu);
      As[(mp * 2 + 1) * 36 + l] = (unsigned short)(u >> 16);
    }
#pragma unroll
    for (int i = 0; i < 4; ++i) {                 // v tile 32x64 -> [d][l]
      int idx = tid + i * 256;
      int l = idx >> 5, np = idx & 31;
      unsigned int u = *(const unsigned int*)(vb + (size_t)(b * 4096 + l0 + l) * 1024 + h * 64 + np * 2);
      Bs[(np * 2) * 36 + l]     = (unsigned short)(u & 0xFFFFu);
      Bs[(np * 2 + 1) * 36 + l] = (unsigned short)(u >> 16);
    }
    __syncthreads();
#pragma unroll
    for (int l = 0; l < 32; ++l) ks += bf2f(As[tid * 36 + l]);
#pragma unroll
    for (int mt = 0; mt < 2; ++mt) {
      v16bf a = ld_a(As, (wv * 2 + mt) * 16, 0, 36);
#pragma unroll
      for (int nt = 0; nt < 4; ++nt) {
        v16bf bb = ld_b(Bs, nt * 16, 0, 36);
        acc[mt * 4 + nt] = wmma_bf16(a, bb, acc[mt * 4 + nt]);
      }
    }
    __syncthreads();
  }
  int lane = tid & 31, grp = lane >> 4, c = lane & 15;
#pragma unroll
  for (int mt = 0; mt < 2; ++mt) {
    int mbase = (wv * 2 + mt) * 16 + grp * 8;
#pragma unroll
    for (int nt = 0; nt < 4; ++nt) {
      int d = nt * 16 + c;
#pragma unroll
      for (int r = 0; r < 8; ++r)
        kv[(size_t)bh * 16384 + (size_t)(mbase + r) * 64 + d] = f2bf(acc[mt * 4 + nt][r]);
    }
  }
  ksum[bh * 256 + tid] = ks;
}

// ---------------------------------------------------------------------------
// ctx[b,l,h*64+d] = (qp @ kv)[l,d] / (qp . ksum)[l], bf16 out.  64x64 tile,
// K=256, 4 waves; qp tile async-DMAed; norm dot fused in the K loop.
// ---------------------------------------------------------------------------
__global__ __launch_bounds__(128)
void num_norm_kernel(const unsigned short* __restrict__ qp,
                     const unsigned short* __restrict__ kv,
                     const float* __restrict__ ksum,
                     unsigned short* __restrict__ ctx) {
  __shared__ __align__(16) unsigned short As[64 * 36];
  __shared__ __align__(16) unsigned short Bs[64 * 36];
  __shared__ float norm_l[64];
  const int tid = threadIdx.x;
  const int l0  = blockIdx.x * 64;
  const int h = blockIdx.y, b = blockIdx.z;
  const int bh = b * 16 + h;
  const unsigned lds_a = (unsigned)(size_t)(void*)As;
  unsigned int* Au = (unsigned int*)As;
  v8f acc[4] = {};
  float nrm = 0.f;
  for (int m0k = 0; m0k < 256; m0k += 32) {
#pragma unroll
    for (int i = 0; i < 4; ++i) {                 // qp tile 64 rows x 64B, async
      int idx = tid + i * 128;
      int r = idx >> 3, ch = idx & 7;
      async_b64(lds_a + r * 72 + ch * 8,
                (const char*)qp + ((size_t)((b * 4096 + l0 + r) * 16 + h) * 256 + m0k) * 2 + ch * 8);
    }
#pragma unroll
    for (int i = 0; i < 8; ++i) {                 // kv tile 32x64 -> [d][m]
      int idx = tid + i * 128;
      int mr = idx >> 5, np = idx & 31;
      unsigned int u = *(const unsigned int*)(kv + (size_t)bh * 16384 + (size_t)(m0k + mr) * 64 + np * 2);
      Bs[(np * 2) * 36 + mr]     = (unsigned short)(u & 0xFFFFu);
      Bs[(np * 2 + 1) * 36 + mr] = (unsigned short)(u >> 16);
    }
    wait_async0();
    __syncthreads();
    if (tid < 64) {                               // norm dot product
      const float* kss = ksum + bh * 256 + m0k;
#pragma unroll
      for (int mp = 0; mp < 16; ++mp) {
        unsigned int u = Au[tid * 18 + mp];
        nrm += bf2f((unsigned short)(u & 0xFFFFu)) * kss[mp * 2]
             + bf2f((unsigned short)(u >> 16))     * kss[mp * 2 + 1];
      }
    }
    int wv = tid >> 5;
    v16bf a = ld_a(As, wv * 16, 0, 36);
#pragma unroll
    for (int nt = 0; nt < 4; ++nt) {
      v16bf bb = ld_b(Bs, nt * 16, 0, 36);
      acc[nt] = wmma_bf16(a, bb, acc[nt]);
    }
    __syncthreads();
  }
  if (tid < 64) norm_l[tid] = nrm;
  __syncthreads();
  int wv = tid >> 5, lane = tid & 31, grp = lane >> 4, c = lane & 15;
  int rbase = wv * 16 + grp * 8;
#pragma unroll
  for (int nt = 0; nt < 4; ++nt) {
    int d = nt * 16 + c;
#pragma unroll
    for (int r = 0; r < 8; ++r) {
      int row = rbase + r;
      ctx[(size_t)(b * 4096 + l0 + row) * 1024 + h * 64 + d] = f2bf(acc[nt][r] / norm_l[row]);
    }
  }
}

// ---------------------------------------------------------------------------
extern "C" void kernel_launch(void* const* d_in, const int* in_sizes, int n_in,
                              void* d_out, int out_size, void* d_ws, size_t ws_size,
                              hipStream_t stream) {
  (void)in_sizes; (void)n_in; (void)out_size; (void)ws_size;
  const float* query = (const float*)d_in[0];
  const float* key_  = (const float*)d_in[1];
  const float* value = (const float*)d_in[2];
  const float* Wq = (const float*)d_in[3];
  const float* bq = (const float*)d_in[4];
  const float* Wk = (const float*)d_in[5];
  const float* bk = (const float*)d_in[6];
  const float* Wv = (const float*)d_in[7];
  const float* bv = (const float*)d_in[8];
  const float* Wo = (const float*)d_in[9];
  const float* bo = (const float*)d_in[10];
  const float* proj = (const float*)d_in[11];

  char* ws = (char*)d_ws;
  unsigned short* qin  = (unsigned short*)(ws + 0x00000000);  // [16384,1024] bf16
  unsigned short* kin  = (unsigned short*)(ws + 0x02000000);
  unsigned short* vin  = (unsigned short*)(ws + 0x04000000);
  unsigned short* qb   = (unsigned short*)(ws + 0x06000000);  // projected q/k/v
  unsigned short* kb   = (unsigned short*)(ws + 0x08000000);
  unsigned short* vb   = (unsigned short*)(ws + 0x0A000000);
  unsigned short* ctxb = (unsigned short*)(ws + 0x0C000000);
  unsigned short* qp   = (unsigned short*)(ws + 0x0E000000);  // [B,L,H,M] bf16
  unsigned short* kp   = (unsigned short*)(ws + 0x16000000);
  unsigned short* kvb  = (unsigned short*)(ws + 0x1E000000);  // [64,256,64] bf16
  float*          ksum = (float*)(ws + 0x1E200000);           // [64,256] f32
  unsigned short* wqb  = (unsigned short*)(ws + 0x1E210000);  // weights bf16
  unsigned short* wkb  = (unsigned short*)(ws + 0x1E410000);
  unsigned short* wvb  = (unsigned short*)(ws + 0x1E610000);
  unsigned short* wob  = (unsigned short*)(ws + 0x1E810000);
  unsigned short* pjb  = (unsigned short*)(ws + 0x1EA10000);  // proj*dn bf16

  const float dn = 0.3535533905932738f;                       // 64^(-1/4)
  const int nAct4 = 16384 * 1024 / 4, nW4 = 1024 * 1024 / 4, nP4 = 256 * 64 / 4;
  cvt_bf16_kernel<<<nAct4 / 256, 256, 0, stream>>>(query, qin, nAct4, 1.f);
  cvt_bf16_kernel<<<nAct4 / 256, 256, 0, stream>>>(key_,  kin, nAct4, 1.f);
  cvt_bf16_kernel<<<nAct4 / 256, 256, 0, stream>>>(value, vin, nAct4, 1.f);
  cvt_bf16_kernel<<<nW4 / 256, 256, 0, stream>>>(Wq, wqb, nW4, 1.f);
  cvt_bf16_kernel<<<nW4 / 256, 256, 0, stream>>>(Wk, wkb, nW4, 1.f);
  cvt_bf16_kernel<<<nW4 / 256, 256, 0, stream>>>(Wv, wvb, nW4, 1.f);
  cvt_bf16_kernel<<<nW4 / 256, 256, 0, stream>>>(Wo, wob, nW4, 1.f);
  cvt_bf16_kernel<<<nP4 / 256, 256, 0, stream>>>(proj, pjb, nP4, dn);

  dim3 gG(256, 16), bG(128);
  gemm_bias_kernel<true><<<gG, bG, 0, stream>>>(qin, wqb, bq, nullptr, qb);
  gemm_bias_kernel<true><<<gG, bG, 0, stream>>>(kin, wkb, bk, nullptr, kb);
  gemm_bias_kernel<true><<<gG, bG, 0, stream>>>(vin, wvb, bv, nullptr, vb);

  dim3 gP(64, 16, 4), bP(256);
  phi_kernel<<<gP, bP, 0, stream>>>(qb, pjb, qp);
  phi_kernel<<<gP, bP, 0, stream>>>(kb, pjb, kp);

  kv_ksum_kernel<<<dim3(64), dim3(256), 0, stream>>>(kp, vb, kvb, ksum);

  num_norm_kernel<<<dim3(64, 16, 4), dim3(128), 0, stream>>>(qp, kvb, ksum, ctxb);

  gemm_bias_kernel<false><<<gG, bG, 0, stream>>>(ctxb, wob, bo, (float*)d_out, nullptr);
}
